// Pointer_55284819034614
// MI455X (gfx1250) — matrix operand order repaired and checked by reading, not compiled, not run on hardware
//
#include <hip/hip_runtime.h>

#define D 2048
#define S 16384
#define HREAL 250
#define HPAD 256

typedef __attribute__((ext_vector_type(16))) __bf16 bf16x16;
typedef __attribute__((ext_vector_type(8)))  float  f32x8;

union FragU { bf16x16 v; uint4 q[2]; };

__device__ __forceinline__ unsigned short f2bf(float f) {
  unsigned int u = __float_as_uint(f);
  unsigned int r = u + 0x7FFFu + ((u >> 16) & 1u);   // round-to-nearest-even
  return (unsigned short)(r >> 16);
}

// fast tanh: 1 - 2/(e^{2x}+1); saturates correctly to +/-1 for large |x|
__device__ __forceinline__ float fast_tanh(float x) {
  const float t = __expf(2.f * x);
  return 1.f - 2.f * __builtin_amdgcn_rcpf(t + 1.f);
}

// A fragment (16x32, 16-bit): lane holds row = lane&15; kBase = (lane<16)?0:8.
__device__ __forceinline__ bf16x16 load_frag_a(const unsigned short* row, int k0, int lane) {
  const int kb = (lane & 16) ? 8 : 0;
  FragU f;
  f.q[0] = *(const uint4*)(row + k0 + kb);
  f.q[1] = *(const uint4*)(row + k0 + 16 + kb);
  return f.v;
}

// B fragment (32x16, 16-bit): lane holds col = lane&15; kBase = (lane<16)?0:8.
__device__ __forceinline__ bf16x16 load_frag_b(const unsigned short* p, int lane) {
  const int kb = (lane & 16) ? 8 : 0;
  FragU f;
  f.q[0] = *(const uint4*)(p + kb);
  f.q[1] = *(const uint4*)(p + kb + 16);
  return f.v;
}

// ---------------- weight conversion kernels ----------------
__global__ void cvt_wctx_kernel(const float* __restrict__ w, unsigned short* __restrict__ o) {
  const size_t g = (size_t)blockIdx.x * 256 + threadIdx.x;      // 4M/4 threads
  const float4 f = *(const float4*)(w + g * 4);
  ushort4 u;
  u.x = f2bf(f.x); u.y = f2bf(f.y); u.z = f2bf(f.z); u.w = f2bf(f.w);
  *(ushort4*)(o + g * 4) = u;
}

__global__ void cvt_wp1_kernel(const float* __restrict__ w, unsigned short* __restrict__ o) {
  const int g = blockIdx.x * 256 + threadIdx.x;                  // 256*2048 elems
  const int row = g >> 11, col = g & (D - 1);
  o[g] = (row < HREAL) ? f2bf(w[(size_t)row * (3 * D) + col]) : (unsigned short)0;
}

// ---------------- twq = tanh(W_q s_prev + b_q), tdh = tanh(W_info cur) ----------------
__global__ __launch_bounds__(256) void matvec_tanh_kernel(
    const float* __restrict__ W_q, const float* __restrict__ s_prev, const float* __restrict__ b_q,
    const float* __restrict__ W_info, const float* __restrict__ cur,
    float* __restrict__ twq, float* __restrict__ tdh) {
  const int lane = threadIdx.x & 31, wave = threadIdx.x >> 5;
  const int row = blockIdx.x * 8 + wave;
  const int which = blockIdx.y;
  const float* Wm = which ? W_info : W_q;
  const float* x  = which ? cur : s_prev;
  float acc = 0.f;
  for (int e = lane; e < D; e += 32) acc = fmaf(Wm[(size_t)row * D + e], x[e], acc);
  #pragma unroll
  for (int off = 16; off > 0; off >>= 1) acc += __shfl_xor(acc, off, 32);
  if (lane == 0) {
    const float r = which ? acc : acc + b_q[row];
    (which ? tdh : twq)[row] = tanhf(r);
  }
}

// ---------------- c12[h'] = b_p1 + W_p1[:,D:2D]·twq + W_p1[:,2D:3D]·tdh ; pad W_p2 ----------------
__global__ __launch_bounds__(256) void c12_kernel(
    const float* __restrict__ W_p1, const float* __restrict__ b_p1, const float* __restrict__ W_p2,
    const float* __restrict__ twq, const float* __restrict__ tdh,
    float* __restrict__ c12, float* __restrict__ w2pad) {
  const int lane = threadIdx.x & 31, wave = threadIdx.x >> 5;
  const int row = blockIdx.x * 8 + wave;                          // 0..255
  float acc = 0.f;
  if (row < HREAL) {
    const float* wr = W_p1 + (size_t)row * (3 * D);
    for (int e = lane; e < D; e += 32)
      acc += wr[D + e] * twq[e] + wr[2 * D + e] * tdh[e];
  }
  #pragma unroll
  for (int off = 16; off > 0; off >>= 1) acc += __shfl_xor(acc, off, 32);
  if (lane == 0) {
    c12[row]   = (row < HREAL) ? acc + b_p1[row] : 0.f;
    w2pad[row] = (row < HREAL) ? W_p2[row] : 0.f;
  }
}

// ---------------- fused scores kernel: M-tile 64, 4 waves, LDS-staged B, SW-pipelined ----------------
#define ENC_STRIDE 2056    // 2048+8 halfwords: 16B aligned, conflict-free A frags
#define BSL_STRIDE 136     // 128+8 halfwords
#define TCH_STRIDE 40      // 16B aligned; 20n mod 64 distinct for n=0..15
#define WPS_STRIDE 40
#define SLAB_E 128
#define NSLAB (D / SLAB_E) // 16

// LDS-bound to 1 workgroup/WGP (1 wave/SIMD) -> give the whole register file to the wave
__global__ __attribute__((amdgpu_waves_per_eu(1))) __launch_bounds__(128)
void attn_scores_kernel(
    const float* __restrict__ enc,
    const unsigned short* __restrict__ wctx,   // [2048][2048] bf16
    const unsigned short* __restrict__ wp1,    // [256][2048] bf16 (rows >=250 zero)
    const float* __restrict__ c12,
    const float* __restrict__ w2,
    float* __restrict__ scores) {
  extern __shared__ unsigned short smem[];
  unsigned short* encT = smem;                                   // 64 x ENC_STRIDE
  unsigned short* bsl  = encT + 64 * ENC_STRIDE;                 // 32 x BSL_STRIDE (e-slab of W_ctx)
  unsigned short* tch  = bsl + 32 * BSL_STRIDE;                  // 64 x TCH_STRIDE
  unsigned short* wps  = tch + 64 * TCH_STRIDE;                  // 256 x WPS_STRIDE (W_p1 d-chunk)

  const int tid  = threadIdx.x;
  const int lane = tid & 31;
  const int wave = tid >> 5;                                     // 4 waves x 16 S-rows
  const int sBase = blockIdx.x * 64;

  // stage 64 enc rows as bf16 into LDS (enc_hs read exactly once from HBM)
  for (int i = tid * 4; i < 64 * D; i += 128 * 4) {
    const int row = i >> 11, col = i & (D - 1);
    const float4 f = *(const float4*)(enc + (size_t)(sBase + row) * D + col);
    ushort4 u;
    u.x = f2bf(f.x); u.y = f2bf(f.y); u.z = f2bf(f.z); u.w = f2bf(f.w);
    *(ushort4*)(encT + row * ENC_STRIDE + col) = u;
  }
  __syncthreads();

  const int n = lane & 15;
  const int hi = (lane & 16) ? 1 : 0;
  const unsigned short* aRow  = encT + (size_t)(wave * 16 + n) * ENC_STRIDE;
  const unsigned short* a2Row = tch  + (size_t)(wave * 16 + n) * TCH_STRIDE;
  const unsigned short* bslRow0 = bsl + (size_t)n * BSL_STRIDE;
  const unsigned short* bslRow1 = bsl + (size_t)(16 + n) * BSL_STRIDE;
  const int mBase = wave * 16 + hi * 8;

  // B-slab staging: 32 rows x 128 cols bf16, 128 threads -> 32 hw each
  const int bRow = tid >> 2;             // 0..31
  const int bSeg = (tid & 3) * 32;       // col offset (halfwords)
  // W_p1 chunk staging: rows 2t, 2t+1, one uint4 in flight (pieces 0..7)
  const int pRow = tid * 2;

  f32x8 hAcc[16];
  #pragma unroll
  for (int t = 0; t < 16; ++t) hAcc[t] = f32x8{0.f,0.f,0.f,0.f,0.f,0.f,0.f,0.f};

  #pragma unroll 1
  for (int d0 = 0; d0 < D; d0 += 32) {
    const unsigned short* pw = wp1 + (size_t)pRow * D + d0;
    uint4 wpiece = {0, 0, 0, 0};

    f32x8 c0 = {0.f,0.f,0.f,0.f,0.f,0.f,0.f,0.f};
    f32x8 c1 = {0.f,0.f,0.f,0.f,0.f,0.f,0.f,0.f};

    // prologue: load e-slab 0 of W_ctx rows [d0, d0+32)
    const unsigned short* bg = wctx + (size_t)(d0 + bRow) * D + bSeg;
    uint4 br[4];
    #pragma unroll
    for (int j = 0; j < 4; ++j) br[j] = *(const uint4*)(bg + 8 * j);

    #pragma unroll 1
    for (int s = 0; s < NSLAB; ++s) {
      __syncthreads();                                   // all waves done reading previous slab
      unsigned short* bq = bsl + bRow * BSL_STRIDE + bSeg;
      #pragma unroll
      for (int j = 0; j < 4; ++j) *(uint4*)(bq + 8 * j) = br[j];
      if (s < NSLAB - 1) {                               // prefetch next slab during compute
        const unsigned short* bg2 = bg + (s + 1) * SLAB_E;
        #pragma unroll
        for (int j = 0; j < 4; ++j) br[j] = *(const uint4*)(bg2 + 8 * j);
      }
      // W_p1 chunk staging spread across slabs: store piece s-1, load piece s
      if (s >= 1 && s <= 8) {
        const int p = s - 1;
        *(uint4*)(wps + (size_t)(pRow + (p >> 2)) * WPS_STRIDE + (p & 3) * 8) = wpiece;
      }
      if (s < 8) {
        wpiece = *(const uint4*)(pw + (s >> 2) * D + (s & 3) * 8);
      }
      __syncthreads();                                   // slab visible

      const int e0 = s * SLAB_E;
      // depth-1 pipeline on a/b0; b1 issued before the first WMMA so its wait overlaps it
      bf16x16 aC  = load_frag_a(aRow, e0, lane);
      bf16x16 b0C = load_frag_b(bslRow0, lane);
      #pragma unroll
      for (int k = 0; k < SLAB_E; k += 32) {
        const bf16x16 b1C = load_frag_b(bslRow1 + k, lane);
        bf16x16 aN = aC, b0N = b0C;
        if (k + 32 < SLAB_E) {
          aN  = load_frag_a(aRow, e0 + k + 32, lane);
          b0N = load_frag_b(bslRow0 + k + 32, lane);
        }
        c0 = __builtin_amdgcn_wmma_f32_16x16x32_bf16(false, aC, false, b0C, (short)0, c0, false, false);
        c1 = __builtin_amdgcn_wmma_f32_16x16x32_bf16(false, aC, false, b1C, (short)0, c1, false, false);
        aC = aN; b0C = b0N;
      }
    }

    // tanh(uh) tile -> tch (wave-exclusive rows)
    #pragma unroll
    for (int i = 0; i < 8; ++i) {
      tch[(mBase + i) * TCH_STRIDE + n]      = f2bf(fast_tanh(c0[i]));
      tch[(mBase + i) * TCH_STRIDE + 16 + n] = f2bf(fast_tanh(c1[i]));
    }
    __syncthreads();

    // GEMM2: h[s,0:256] += tanh(uh_chunk) @ W_p1_chunk^T, operands from LDS, pipelined
    const bf16x16 a2 = load_frag_a(a2Row, 0, lane);
    bf16x16 bC = load_frag_b(wps + (size_t)n * WPS_STRIDE, lane);
    #pragma unroll
    for (int t = 0; t < 16; ++t) {
      bf16x16 bN = bC;
      if (t < 15) bN = load_frag_b(wps + (size_t)((t + 1) * 16 + n) * WPS_STRIDE, lane);
      hAcc[t] = __builtin_amdgcn_wmma_f32_16x16x32_bf16(false, a2, false, bC, (short)0, hAcc[t], false, false);
      bC = bN;
    }
    __syncthreads();                                     // protect wps/tch for next iteration
  }

  // + c12, relu, dot with W_p2 -> per-row score
  float rp[8] = {0.f,0.f,0.f,0.f,0.f,0.f,0.f,0.f};
  #pragma unroll
  for (int t = 0; t < 16; ++t) {
    const int col = t * 16 + n;
    const float cc = c12[col];
    const float wv = w2[col];
    #pragma unroll
    for (int i = 0; i < 8; ++i) {
      float h = hAcc[t][i] + cc;
      h = fmaxf(h, 0.f);
      rp[i] = fmaf(h, wv, rp[i]);
    }
  }
  #pragma unroll
  for (int off = 1; off < 16; off <<= 1) {
    #pragma unroll
    for (int i = 0; i < 8; ++i) rp[i] += __shfl_xor(rp[i], off, 32);
  }
  if (n == 0) {
    #pragma unroll
    for (int i = 0; i < 8; ++i) scores[sBase + mBase + i] = rp[i];
  }
}

// ---------------- masked softmax over S (single workgroup) ----------------
__global__ __launch_bounds__(1024) void softmax_kernel(
    const float* __restrict__ scores, const int* __restrict__ mask, float* __restrict__ attn) {
  __shared__ float red[32];
  const int tid = threadIdx.x, lane = tid & 31, wid = tid >> 5;
  const float NEG_INF = __int_as_float((int)0xff800000);
  float v[16];
  float lmax = NEG_INF;
  #pragma unroll
  for (int i = 0; i < 16; ++i) {
    const int s = i * 1024 + tid;
    float x = scores[s];
    if (mask[s] != 0) x = NEG_INF;
    v[i] = x;
    lmax = fmaxf(lmax, x);
  }
  #pragma unroll
  for (int off = 16; off > 0; off >>= 1) lmax = fmaxf(lmax, __shfl_xor(lmax, off, 32));
  if (lane == 0) red[wid] = lmax;
  __syncthreads();
  if (tid < 32) {
    float m = red[tid];
    #pragma unroll
    for (int off = 16; off > 0; off >>= 1) m = fmaxf(m, __shfl_xor(m, off, 32));
    if (tid == 0) red[0] = m;
  }
  __syncthreads();
  const float bmax = red[0];
  __syncthreads();
  float lsum = 0.f;
  #pragma unroll
  for (int i = 0; i < 16; ++i) { const float e = expf(v[i] - bmax); v[i] = e; lsum += e; }
  #pragma unroll
  for (int off = 16; off > 0; off >>= 1) lsum += __shfl_xor(lsum, off, 32);
  if (lane == 0) red[wid] = lsum;
  __syncthreads();
  if (tid < 32) {
    float sm = red[tid];
    #pragma unroll
    for (int off = 16; off > 0; off >>= 1) sm += __shfl_xor(sm, off, 32);
    if (tid == 0) red[0] = sm;
  }
  __syncthreads();
  const float inv = 1.f / red[0];
  #pragma unroll
  for (int i = 0; i < 16; ++i) attn[i * 1024 + tid] = v[i] * inv;
}

// ---------------- ctx = sum_s attn[s] * enc[s,:], deterministic two-stage ----------------
__global__ void ctx_partial_kernel(const float* __restrict__ enc, const float* __restrict__ attn,
                                   float* __restrict__ part) {
  const int d = blockIdx.x * 256 + threadIdx.x;
  const int s0 = blockIdx.y * 512;
  float acc = 0.f;
  for (int s = s0; s < s0 + 512; ++s)
    acc = fmaf(enc[(size_t)s * D + d], attn[s], acc);
  part[(size_t)blockIdx.y * D + d] = acc;
}

__global__ void ctx_reduce_kernel(const float* __restrict__ part, float* __restrict__ ctx) {
  const int d = blockIdx.x * 256 + threadIdx.x;
  float acc = 0.f;
  #pragma unroll
  for (int j = 0; j < 32; ++j) acc += part[(size_t)j * D + d];
  ctx[d] = acc;
}

// ---------------- host ----------------
extern "C" void kernel_launch(void* const* d_in, const int* in_sizes, int n_in,
                              void* d_out, int out_size, void* d_ws, size_t ws_size,
                              hipStream_t stream) {
  const float* s_prev = (const float*)d_in[0];
  const float* enc    = (const float*)d_in[1];
  const float* cur    = (const float*)d_in[2];
  const int*   mask   = (const int*)d_in[3];
  const float* W_info = (const float*)d_in[4];
  const float* W_ctx  = (const float*)d_in[5];
  const float* W_q    = (const float*)d_in[6];
  const float* b_q    = (const float*)d_in[7];
  const float* W_p1   = (const float*)d_in[8];
  const float* b_p1   = (const float*)d_in[9];
  const float* W_p2   = (const float*)d_in[10];
  // d_in[11] = b_p2: shift-invariant under softmax; attn and ctx do not depend on it.

  float* ws      = (float*)d_ws;
  float* twq     = ws + 0;        // 2048
  float* tdh     = ws + 2048;     // 2048
  float* c12     = ws + 4096;     // 256
  float* w2      = ws + 4352;     // 256
  float* scores  = ws + 4608;     // 16384
  float* part    = ws + 20992;    // 32*2048
  unsigned short* wctx_bf = (unsigned short*)((char*)d_ws + 346112);   // 2048*2048 bf16 (8MB)
  unsigned short* wp1_bf  = (unsigned short*)((char*)d_ws + 8734720);  // 256*2048 bf16 (1MB)

  float* attn = (float*)d_out;
  float* ctx  = (float*)d_out + S;

  cvt_wctx_kernel<<<4096, 256, 0, stream>>>(W_ctx, wctx_bf);
  cvt_wp1_kernel<<<2048, 256, 0, stream>>>(W_p1, wp1_bf);
  matvec_tanh_kernel<<<dim3(256, 2), 256, 0, stream>>>(W_q, s_prev, b_q, W_info, cur, twq, tdh);
  c12_kernel<<<32, 256, 0, stream>>>(W_p1, b_p1, W_p2, twq, tdh, c12, w2);

  const size_t smem = (size_t)(64 * ENC_STRIDE + 32 * BSL_STRIDE +
                               64 * TCH_STRIDE + 256 * WPS_STRIDE) * sizeof(unsigned short);
  attn_scores_kernel<<<256, 128, smem, stream>>>(enc, wctx_bf, wp1_bf, c12, w2, scores);

  softmax_kernel<<<1, 1024, 0, stream>>>(scores, mask, attn);
  ctx_partial_kernel<<<dim3(8, 32), 256, 0, stream>>>(enc, attn, part);
  ctx_reduce_kernel<<<8, 256, 0, stream>>>(part, ctx);
}
